// RTM_25615184953813
// MI455X (gfx1250) — matrix operand order repaired and checked by reading, not compiled, not run on hardware
//
#include <hip/hip_runtime.h>
#include <hip/hip_bf16.h>

typedef __attribute__((ext_vector_type(16))) _Float16 v16h;
typedef __attribute__((ext_vector_type(8)))  _Float16 v8h;
typedef __attribute__((ext_vector_type(8)))  float    v8f;
typedef __attribute__((ext_vector_type(4)))  int      v4i;
typedef unsigned int u32;

#define ROWS 65536   // L*N
#define MDIM 256
#define NSEQ 512
#define LDIM 128

// ---------------------------------------------------------------------------
// CDNA5 async global->LDS staging (ASYNCcnt-tracked), with sync fallback.
// Builtin signature (from clang diagnostic): (v4i addrspace(1)*, v4i
// addrspace(3)*, imm offset, imm cpol).
// ---------------------------------------------------------------------------
#if defined(__AMDGCN__) && __has_builtin(__builtin_amdgcn_global_load_async_to_lds_b128)
#define HAVE_ASYNC 1
#else
#define HAVE_ASYNC 0
#endif

typedef __attribute__((address_space(1))) v4i gv4i;
typedef __attribute__((address_space(3))) v4i lv4i;

__device__ __forceinline__ void async_b128(const _Float16* g, _Float16* l) {
#if HAVE_ASYNC
  __builtin_amdgcn_global_load_async_to_lds_b128(
      (gv4i*)(size_t)(const void*)g, (lv4i*)l, 0, 0);
#else
  *(uint4*)l = *(const uint4*)g;
#endif
}

__device__ __forceinline__ void wait_async0() {
#if HAVE_ASYNC
#if __has_builtin(__builtin_amdgcn_s_wait_asynccnt)
  __builtin_amdgcn_s_wait_asynccnt(0);
#else
  asm volatile("s_wait_asynccnt 0x0" ::: "memory");
#endif
#endif
}

// ---------------------------------------------------------------------------
// fragment loaders (wave32, 16x16x32 f16 WMMA layouts per CDNA5 ISA 7.12.2)
// A (16x32, row-major [m][k]): lanes 0-15 hold K=0..7 & 16..23 (pairs),
//                              lanes 16-31 hold K=8..15 & 24..31.
// B (32x16, stored as [n][k]): same K order as A (lanes 0-15 K=0..7 & 16..23).
// ---------------------------------------------------------------------------
__device__ __forceinline__ v16h frag_A_lds(const _Float16* As, int r0) {
  int lane = threadIdx.x & 31;
  int m = lane & 15;
  int kb = (lane >> 4) << 3;            // 0 or 8
  const _Float16* p = As + (size_t)(r0 + m) * 40;
  v16h f;
#pragma unroll
  for (int i = 0; i < 4; ++i) {
    f[2 * i]     = p[kb + 2 * i];
    f[2 * i + 1] = p[kb + 2 * i + 1];
    f[8 + 2 * i] = p[kb + 16 + 2 * i];
    f[9 + 2 * i] = p[kb + 17 + 2 * i];
  }
  return f;
}

__device__ __forceinline__ v16h frag_B_lds(const _Float16* Bs, int n0) {
  int lane = threadIdx.x & 31;
  int n = lane & 15;
  int kb = (lane >> 4) << 3;            // 0 or 8
  const _Float16* p = Bs + (size_t)(n0 + n) * 40 + kb;
  v16h f;
#pragma unroll
  for (int i = 0; i < 8; ++i) {
    f[2 * i]     = p[2 * i + (i >= 4 ? 8 : 0)];
    f[2 * i + 1] = p[2 * i + 1 + (i >= 4 ? 8 : 0)];
  }
  return f;
}

__device__ __forceinline__ v8f wmma_f16(v16h a, v16h b, v8f c) {
  return __builtin_amdgcn_wmma_f32_16x16x32_f16(
      /*neg_a=*/false, a, /*neg_b=*/false, b,
      /*c_mod=*/(short)0, c, /*reuse_a=*/false, /*reuse_b=*/false);
}

// issue one 64x32 f16 tile pair (A rows / B out-cols) as 16B async transfers
__device__ __forceinline__ void stage(_Float16* as, _Float16* bs,
                                      const _Float16* A, const _Float16* B,
                                      int K, int row0, int col0, int k0,
                                      int tid) {
#pragma unroll
  for (int t = 0; t < 2; ++t) {
    int d = tid + t * 128;              // 0..255 transfers of 16B
    int r = d >> 2;                     // row within tile
    int q = (d & 3) * 8;                // 8-half chunk within row
    async_b128(A + (size_t)(row0 + r) * K + k0 + q, as + r * 40 + q);
    async_b128(B + (size_t)(col0 + r) * K + k0 + q, bs + r * 40 + q);
  }
}

// ---------------------------------------------------------------------------
// Tiled WMMA GEMM:  out[row, col] = sum_k A[row,k] * B[col,k]  (+ epilogue)
// Tile 64x64 per WG, 4 waves, 32x32 per wave (2x2 fragments), K-step 32,
// double-buffered LDS with async global->LDS staging.
// ---------------------------------------------------------------------------
enum { M_POSBIAS = 0, M_F16 = 1, M_RES = 2, M_GELU = 3, M_BIAS = 4 };

template <int MODE>
__global__ __launch_bounds__(128) void gemm_kernel(
    const _Float16* __restrict__ A, const _Float16* __restrict__ B, int K,
    int cols, const float* __restrict__ extra, float* __restrict__ outF,
    _Float16* __restrict__ outH) {
  __shared__ _Float16 As[2][64 * 40];
  __shared__ _Float16 Bs[2][64 * 40];
  const int row0 = blockIdx.y * 64;
  const int col0 = blockIdx.x * 64;
  const int tid  = threadIdx.x;
  const int lane = tid & 31;
  const int wave = tid >> 5;
  const int wr = (wave >> 1) * 32;      // wave row offset in tile
  const int wc = (wave & 1) * 32;       // wave col offset in tile

  v8f c[2][2] = {};

  stage(As[0], Bs[0], A, B, K, row0, col0, 0, tid);

  const int steps = K >> 5;
  for (int s = 0; s < steps; ++s) {
    wait_async0();                      // own async ops for tile s landed
    __syncthreads();                    // all waves' data in; prev readers done
    if (s + 1 < steps)
      stage(As[(s + 1) & 1], Bs[(s + 1) & 1], A, B, K, row0, col0,
            (s + 1) << 5, tid);         // overlap next copies with compute
    const _Float16* a_ = As[s & 1];
    const _Float16* b_ = Bs[s & 1];
    v16h a0 = frag_A_lds(a_, wr);
    v16h a1 = frag_A_lds(a_, wr + 16);
    v16h b0 = frag_B_lds(b_, wc);
    v16h b1 = frag_B_lds(b_, wc + 16);
    c[0][0] = wmma_f16(a0, b0, c[0][0]);
    c[0][1] = wmma_f16(a0, b1, c[0][1]);
    c[1][0] = wmma_f16(a1, b0, c[1][0]);
    c[1][1] = wmma_f16(a1, b1, c[1][1]);
  }

  // epilogue: C/D layout -> lane n = lane&15, m = (lane>>4)*8 + r
  const int n = lane & 15;
  const int mb = (lane >> 4) * 8;
#pragma unroll
  for (int i = 0; i < 2; ++i) {
#pragma unroll
    for (int jj = 0; jj < 2; ++jj) {
      int col = col0 + wc + jj * 16 + n;
#pragma unroll
      for (int r = 0; r < 8; ++r) {
        size_t row = (size_t)(row0 + wr + i * 16 + mb + r);
        size_t off = row * (size_t)cols + col;
        float v = c[i][jj][r];
        if (MODE == M_POSBIAS) {        // extra = precomputed posbias[N][M]
          outF[off] = v + extra[(row & (NSEQ - 1)) * (size_t)cols + col];
        } else if (MODE == M_F16) {
          outH[off] = (_Float16)v;
        } else if (MODE == M_RES) {
          outF[off] = v + extra[off];
        } else if (MODE == M_GELU) {
          v += extra[col];
          v = 0.5f * v * (1.0f + erff(v * 0.70710678f));
          outH[off] = (_Float16)v;
        } else {  // M_BIAS
          outF[off] = v + extra[col];
        }
      }
    }
  }
}

// ---------------------------------------------------------------------------
// positional bias table: pb[j, c] = (c odd ? cos : sin)(j / 10000^(2*(c&~1)/M))
// ---------------------------------------------------------------------------
__global__ __launch_bounds__(256) void posbias_kernel(float* __restrict__ pb) {
  int idx = blockIdx.x * 256 + threadIdx.x;  // j*256 + c
  int j = idx >> 8;
  int c = idx & 255;
  float zval = (float)(c & ~1);
  float ang = (float)j * expf(zval * -0.071955804f);  // -2*ln(1e4)/256
  pb[idx] = (c & 1) ? cosf(ang) : sinf(ang);
}

// ---------------------------------------------------------------------------
// LayerNorm over M=256; one wave per row (wave32: 8 cols/lane).
// ADDRES=0: outH = f16(ln(x)).  ADDRES=1: outF = ln(x)+x, outH = f16(same).
// ---------------------------------------------------------------------------
template <int ADDRES>
__global__ __launch_bounds__(256) void ln_kernel(
    const float* __restrict__ in, const float* __restrict__ g,
    const float* __restrict__ b, float* __restrict__ outF,
    _Float16* __restrict__ outH) {
  const int wave = threadIdx.x >> 5;
  const int lane = threadIdx.x & 31;
  const size_t row = (size_t)blockIdx.x * 8 + wave;
  const float* p = in + row * MDIM + lane * 8;
  float4 x0 = ((const float4*)p)[0];
  float4 x1 = ((const float4*)p)[1];
  float vals[8] = {x0.x, x0.y, x0.z, x0.w, x1.x, x1.y, x1.z, x1.w};
  float s = 0.f, q = 0.f;
#pragma unroll
  for (int e = 0; e < 8; ++e) { s += vals[e]; q += vals[e] * vals[e]; }
#pragma unroll
  for (int m = 16; m > 0; m >>= 1) {
    s += __shfl_xor(s, m, 32);
    q += __shfl_xor(q, m, 32);
  }
  const float mean = s * (1.0f / MDIM);
  const float var = q * (1.0f / MDIM) - mean * mean;
  const float rstd = rsqrtf(var + 1e-5f);
  const int c0 = lane * 8;
#pragma unroll
  for (int e = 0; e < 8; ++e) {
    float y = (vals[e] - mean) * rstd * g[c0 + e] + b[c0 + e];
    if (ADDRES) {
      float o = y + vals[e];
      outF[row * MDIM + c0 + e] = o;
      outH[row * MDIM + c0 + e] = (_Float16)o;
    } else {
      outH[row * MDIM + c0 + e] = (_Float16)y;
    }
  }
}

// ---------------------------------------------------------------------------
// rsa[row,h] = (q.k)/sqrt(DH); imv[row, h*32+d] = rsa * v  (qkv f16 [row][768])
// ---------------------------------------------------------------------------
__global__ __launch_bounds__(256) void rsa_kernel(
    const _Float16* __restrict__ qkv, float* __restrict__ imv) {
  size_t idx = (size_t)blockIdx.x * 256 + threadIdx.x;
  size_t row = idx >> 3;
  int h = (int)(idx & 7);
  const v8h* q8 = (const v8h*)(qkv + row * 768 + h * 32);
  float dot = 0.f;
#pragma unroll
  for (int t = 0; t < 4; ++t) {
    v8h qa = q8[t];        // q
    v8h ka = q8[32 + t];   // k (+256 halves)
#pragma unroll
    for (int e = 0; e < 8; ++e) dot += (float)qa[e] * (float)ka[e];
  }
  float r = dot * 0.17677669529f;  // 1/sqrt(32)
  float* o = imv + row * MDIM + h * 32;
#pragma unroll
  for (int t = 0; t < 4; ++t) {
    v8h va = q8[64 + t];   // v (+512 halves)
#pragma unroll
    for (int e = 0; e < 8; ++e) o[t * 8 + e] = r * (float)va[e];
  }
}

// ---------------------------------------------------------------------------
// inclusive cumsum over j (N=512) per (i, m); f32 scan state, f16 output.
// thread-per-column: 128 blocks x 256 threads, fully coalesced.
// ---------------------------------------------------------------------------
__global__ __launch_bounds__(256) void cumsum_kernel(
    const float* __restrict__ in, _Float16* __restrict__ out) {
  const int i = blockIdx.x;
  const int m = threadIdx.x;
  const float* p = in + (size_t)i * NSEQ * MDIM + m;
  _Float16* o = out + (size_t)i * NSEQ * MDIM + m;
  float acc = 0.f;
  for (int j = 0; j < NSEQ; ++j) {
    acc += p[(size_t)j * MDIM];
    o[(size_t)j * MDIM] = (_Float16)acc;
  }
}

// f32 -> f16 conversion
__global__ __launch_bounds__(256) void cvt_kernel(
    const float* __restrict__ in, _Float16* __restrict__ out, int n) {
  int i = blockIdx.x * 256 + threadIdx.x;
  if (i < n) out[i] = (_Float16)in[i];
}

// ---------------------------------------------------------------------------
extern "C" void kernel_launch(void* const* d_in, const int* in_sizes, int n_in,
                              void* d_out, int out_size, void* d_ws,
                              size_t ws_size, hipStream_t stream) {
  const float* x      = (const float*)d_in[0];
  const float* weight = (const float*)d_in[1];
  const float* Wqkv   = (const float*)d_in[2];
  const float* Wo     = (const float*)d_in[3];
  const float* ln_g   = (const float*)d_in[4];
  const float* ln_b   = (const float*)d_in[5];
  const float* lnz_g  = (const float*)d_in[6];
  const float* lnz_b  = (const float*)d_in[7];
  const float* fc1_w  = (const float*)d_in[8];
  const float* fc1_b  = (const float*)d_in[9];
  const float* fc2_w  = (const float*)d_in[10];
  const float* fc2_b  = (const float*)d_in[11];
  float* out = (float*)d_out;

  char* ws = (char*)d_ws;
  float*     S   = (float*)ws;                              // 64 MB fp32 s
  _Float16*  X16 = (_Float16*)(ws + ((size_t)64 << 20));    // 32 MB f16 acts
  _Float16*  BIG = (_Float16*)(ws + ((size_t)96 << 20));    // 128 MB f16 qkv/h
  float*     IMV = (float*)(ws + ((size_t)224 << 20));      // 64 MB fp32 imv
  _Float16*  W16 = (_Float16*)(ws + ((size_t)288 << 20));   // f16 weights
  float*     PB  = (float*)(ws + ((size_t)296 << 20));      // 512 KB posbias
  _Float16* w_weight = W16;                 // 256*256
  _Float16* w_qkv    = w_weight + 65536;    // 4*768*256
  _Float16* w_o      = w_qkv + 786432;      // 4*256*256
  _Float16* w_fc1    = w_o + 262144;        // 1024*256
  _Float16* w_fc2    = w_fc1 + 262144;      // 256*1024

  // convert inputs/weights to f16 GEMM operands; build posbias table
  cvt_kernel<<<(16777216 + 255) / 256, 256, 0, stream>>>(x, X16, 16777216);
  cvt_kernel<<<(65536 + 255) / 256, 256, 0, stream>>>(weight, w_weight, 65536);
  cvt_kernel<<<(786432 + 255) / 256, 256, 0, stream>>>(Wqkv, w_qkv, 786432);
  cvt_kernel<<<(262144 + 255) / 256, 256, 0, stream>>>(Wo, w_o, 262144);
  cvt_kernel<<<(262144 + 255) / 256, 256, 0, stream>>>(fc1_w, w_fc1, 262144);
  cvt_kernel<<<(262144 + 255) / 256, 256, 0, stream>>>(fc2_w, w_fc2, 262144);
  posbias_kernel<<<(NSEQ * MDIM) / 256, 256, 0, stream>>>(PB);

  const dim3 blk(128);
#define GEMM(MODE, Ap, Bp, Kd, COLS, extra, of, oh)                         \
  gemm_kernel<MODE><<<dim3((COLS) / 64, ROWS / 64), blk, 0, stream>>>(      \
      (Ap), (Bp), (Kd), (COLS), (extra), (of), (oh))

  // s = x @ weight^T + pos_bias
  GEMM(M_POSBIAS, X16, w_weight, 256, 256, PB, S, (_Float16*)nullptr);

  for (int a = 0; a < 4; ++a) {
    // ln(s) -> X16 (f16)
    ln_kernel<0><<<ROWS / 8, 256, 0, stream>>>(S, ln_g, ln_b, (float*)nullptr,
                                               X16);
    // qkv = ln(s) @ Wqkv[a]^T -> BIG (f16, [row][768])
    GEMM(M_F16, X16, w_qkv + (size_t)a * 768 * 256, 256, 768,
         (const float*)nullptr, (float*)nullptr, BIG);
    // rsa * v -> IMV (f32)
    rsa_kernel<<<(ROWS * 8) / 256, 256, 0, stream>>>(BIG, IMV);
    // cumsum over sequence -> X16 (f16)
    cumsum_kernel<<<LDIM, 256, 0, stream>>>(IMV, X16);
    // s_new = imv @ Wo[a]^T + s  (element-wise read->write, safe in place)
    GEMM(M_RES, X16, w_o + (size_t)a * 65536, 256, 256, S, S,
         (_Float16*)nullptr);
    // s = ln(s_new) + s_new -> S (f32) and X16 (f16)
    ln_kernel<1><<<ROWS / 8, 256, 0, stream>>>(S, lnz_g, lnz_b, S, X16);
    // h = gelu(s @ fc1^T + b1) -> BIG (f16, [row][1024])
    GEMM(M_GELU, X16, w_fc1, 256, 1024, fc1_b, (float*)nullptr, BIG);
    // s = h @ fc2^T + b2 -> S (or d_out on last block)
    GEMM(M_BIAS, BIG, w_fc2, 1024, 256, fc2_b, (a == 3) ? out : S,
         (_Float16*)nullptr);
  }
#undef GEMM
  (void)in_sizes; (void)n_in; (void)out_size; (void)ws_size;
}
